// AggregPolicy_85169201480382
// MI455X (gfx1250) — matrix operand order repaired and checked by reading, not compiled, not run on hardware
//
#include <hip/hip_runtime.h>

typedef __attribute__((ext_vector_type(16))) _Float16 v16h;
typedef __attribute__((ext_vector_type(8)))  float    v8f;

union HV {
  v16h h;
  unsigned long long q[4];
};

// exp2 that is safe in both compilation passes: amdgcn builtin on device,
// inert stub on the host pass (host never executes device code).
__device__ __forceinline__ float fexp2(float v) {
#if defined(__HIP_DEVICE_COMPILE__) && __has_builtin(__builtin_amdgcn_exp2f)
  return __builtin_amdgcn_exp2f(v);
#elif defined(__HIP_DEVICE_COMPILE__)
  return exp2f(v);
#else
  return v;  // host-pass stub, never called
#endif
}

// Fast transcendentals with pre-folded log2(e) constants:
// v_mul + v_exp_f32 + v_add + v_rcp_f32 (+v_fma for tanh).
__device__ __forceinline__ float fsigmoid(float x) {
  return __builtin_amdgcn_rcpf(1.0f + fexp2(x * -1.442695040888963f));
}
__device__ __forceinline__ float ftanh(float x) {
  // tanh(x) = 1 - 2/(1+exp2(2*log2e*x)); overflow -> rcp(inf)=0 -> +/-1.
  return 1.0f - 2.0f * __builtin_amdgcn_rcpf(1.0f + fexp2(x * 2.885390081777927f));
}

// Cross-lane gathers as DPP (VALU) instead of ds_bpermute (LDS pipe).
// xor 8 within each 16-lane row == row_ror:8  ((l+8)%16 == l^8).
__device__ __forceinline__ float lane_xor8(float v) {
  return __builtin_bit_cast(float,
      __builtin_amdgcn_update_dpp(0, __builtin_bit_cast(int, v),
                                  0x128 /*row_ror:8*/, 0xf, 0xf, true));
}
// xor 4 within each 8-lane group == DPP8 with sel[i] = i^4.
__device__ __forceinline__ float lane_xor4(float v) {
  constexpr int sel = 4 | (5 << 3) | (6 << 6) | (7 << 9) |
                      (0 << 12) | (1 << 15) | (2 << 18) | (3 << 21);
  return __builtin_bit_cast(float,
      __builtin_amdgcn_mov_dpp8(__builtin_bit_cast(int, v), sel));
}

// One wave = 16 samples. States (7 joint GRU states + 1 mixer state), each
// 16 samples x 4 features, in per-wave f16 LDS tiles (row layout). States
// 8/9 are constant rows: C0 = zeros, C1 = {1.0,0,0,0} (bias K-row), so the
// steady-state A build is 16 bare ds_load_b64 from precomputed addresses.
// Per step per wave: 8x V_WMMA_F32_16X16X32_F16.
//   A (16x32 f16): lanes 0-15 K0..7 = [left(4)|right(4)];
//                  lanes 16-31 K8..11 = own h, K12 = 1.0 (bias row).
//   B (32x16 f16): cols 0-3 z pre-act (fused), 4-7 r pre-act (fused),
//                  8-11 i_n, 12-15 h_n; elems 8..15 (K16..31) = 0.
//   C/D (16x16 f32): lane = column, element v = sample row v + 8*(lane>=16).
__global__ void __launch_bounds__(256)
aggreg_policy_wmma(const float* __restrict__ x,
                   const float* __restrict__ Wj,   const float* __restrict__ bj,
                   const float* __restrict__ Wm,   const float* __restrict__ bm,
                   const float* __restrict__ Wih_j, const float* __restrict__ Whh_j,
                   const float* __restrict__ bih_j, const float* __restrict__ bhh_j,
                   const float* __restrict__ Wih_m, const float* __restrict__ Whh_m,
                   const float* __restrict__ bih_m, const float* __restrict__ bhh_m,
                   const float* __restrict__ Wact,  const float* __restrict__ bact,
                   float* __restrict__ out, int Bn)
{
  // [wave][state: 0..6 = h_j[t], 7 = h_m, 8 = C0, 9 = C1][sample m][feature f]
  __shared__ __align__(16) _Float16 H[8][10][16][4];
  __shared__ __align__(16) _Float16 T[8][528];   // store-trash for non-owner lanes

  const int tid  = (int)threadIdx.x;
  const int w    = tid >> 5;
  const int lane = tid & 31;
  const int half = lane >> 4;   // 0: lower 16 lanes, 1: upper
  const int n    = lane & 15;   // B/C column role, and A/LDS row role
  const long s   = (long)blockIdx.x * 128 + (long)w * 16 + n;

  const _Float16* rowb = &H[w][0][n][0];          // row(t) = rowb + t*64 halves

  // ---------------- Build the two B (weight) fragments (one-time) --------
  HV Bj, Bm;
  Bj.q[0] = Bj.q[1] = Bj.q[2] = Bj.q[3] = 0ull;
  Bm.q[0] = Bm.q[1] = Bm.q[2] = Bm.q[3] = 0ull;
  {                                    // half h holds K8h..8h+7; elems 8..15 (K16..31) == 0
    const int  g     = (n < 4) ? n + 4 : (n < 8) ? n - 4 : (n < 12) ? n : n - 4;
    const bool useIH = (n < 12);                  // z,r fused + i_n columns
    const bool useHH = (n < 8) || (n >= 12);      // z,r fused + h_n columns
    #pragma unroll
    for (int e = 0; e < 8; ++e) {      // element e <-> K = 8*half + e for column n
      const int k = 8 * half + e;
      float vj = 0.0f, vm = 0.0f;
      if (k < 8)       { if (useIH) vj = Wih_j[g * 8 + k]; }
      else if (k < 12) { if (useHH) vj = Whh_j[g * 4 + (k - 8)]; }
      else if (k == 12)  vj = (useIH ? bih_j[g] : 0.0f) + (useHH ? bhh_j[g] : 0.0f);
      if (k < 4)       { if (useIH) vm = Wih_m[g * 4 + k]; }
      else if (k < 8)  { if (useHH) vm = Whh_m[g * 4 + (k - 4)]; }
      else if (k == 8)   vm = (useIH ? bih_m[g] : 0.0f) + (useHH ? bhh_m[g] : 0.0f);
      Bj.h[e] = (_Float16)vj;
      Bm.h[e] = (_Float16)vm;
    }
  }

  // ---------------- Per-sample init (lanes 0..15, sample m = n) ----------
  if (half == 0) {
    *(unsigned long long*)&H[w][8][n][0] = 0ull;       // C0: zero row
    *(unsigned long long*)&H[w][9][n][0] = 0x3C00ull;  // C1: {1.0h,0,0,0}
    if (s < (long)Bn) {
      const float* xs = x + s * 19;
      float obs[5], jv[7], jd[7];
      #pragma unroll
      for (int k = 0; k < 5; ++k) obs[k] = xs[k];
      #pragma unroll
      for (int k = 0; k < 7; ++k) { jv[k] = xs[5 + k]; jd[k] = xs[12 + k]; }

      #pragma unroll
      for (int f = 0; f < 4; ++f) {    // h_m = obs @ Wm^T + bm
        float acc = bm[f];
        #pragma unroll
        for (int k = 0; k < 5; ++k) acc += Wm[f * 5 + k] * obs[k];
        H[w][7][n][f] = (_Float16)acc;
      }
      #pragma unroll
      for (int t = 0; t < 7; ++t) {    // h_j[t] = [j,jd] @ Wj^T + bj
        #pragma unroll
        for (int f = 0; f < 4; ++f) {
          float acc = Wj[f * 2 + 0] * jv[t] + Wj[f * 2 + 1] * jd[t] + bj[f];
          H[w][t][n][f] = (_Float16)acc;
        }
      }
    }
  }
  __syncthreads();

  // -------- Loop-invariant per-lane A-load addresses (pure loads in loop) --
  const unsigned long long* AQ0[8];
  const unsigned long long* AQ1[8];
  #pragma unroll
  for (int t = 0; t < 7; ++t) {
    const int tl = (t == 0) ? 7 : t - 1;             // left neighbor row
    AQ0[t] = (const unsigned long long*)(rowb + (half ? t : tl) * 64);
    AQ1[t] = (const unsigned long long*)(rowb + (half ? 9 : ((t < 6) ? t + 1 : 8)) * 64);
  }
  AQ0[7] = (const unsigned long long*)(rowb + (half ? 9 : 0) * 64);  // mixer
  AQ1[7] = (const unsigned long long*)(rowb + (half ? 8 : 7) * 64);

  // h_old read base (all lanes; only lanes 8..11 own the value).
  const _Float16* hb = &H[w][0][0][n & 3];           // elem(t,m) = hb + t*64 + m*4
  // Store pointer: owners write the real slot, others a per-lane trash strip.
  const bool upd = ((n >> 2) == 2);                  // lanes 8..11
  _Float16* hbW = upd ? &H[w][0][0][n & 3] : &T[w][n];

  // A fragments: constant zero halves hoisted out of the recurrence loop.
  HV A[8];
  #pragma unroll
  for (int t = 0; t < 8; ++t) { A[t].q[2] = 0ull; A[t].q[3] = 0ull; }

  // ---------------- 7 recurrence steps ----------------
  for (int it = 0; it < 7; ++it) {
    // A build: 16 bare ds_load_b64 from the OLD state (+ constant rows).
    #pragma unroll
    for (int t = 0; t < 8; ++t) {
      A[t].q[0] = *AQ0[t];
      A[t].q[1] = *AQ1[t];
    }

    // 8 WMMAs on the old state.
    v8f C[8];
    #pragma unroll
    for (int t = 0; t < 8; ++t) {
      v8f cz = {};
      C[t] = __builtin_amdgcn_wmma_f32_16x16x32_f16(
          false, A[t].h, false, (t == 7) ? Bm.h : Bj.h,
          (short)0, cz, false, false);
    }

    // Elementwise GRU update: branch-free, 2 DPP gathers per element.
    #pragma unroll
    for (int t = 0; t < 8; ++t) {
      #pragma unroll
      for (int v = 0; v < 8; ++v) {
        const int m = v + 8 * half;                // sample row of this element
        const float c  = C[t][v];
        const float sg = fsigmoid(c);              // z at lanes 0-3, r at 4-7
        const float s1 = lane_xor8(sg);            // lane 8+f<-z, lane 12+f<-r
        const float pr = c * s1;                   // lane 12+f: h_n * r
        const float s2 = lane_xor4(pr);            // lane 8+f <- r*h_n
        const float nv = ftanh(c + s2);            // lane 8+f: n-gate
        const float ho = (float)hb[t * 64 + m * 4];
        const float hnew = s1 * (ho - nv) + nv;    // (1-z)*n + z*h_old
        hbW[t * 64 + m * 4] = (_Float16)hnew;      // owners -> state, rest -> trash
      }
    }
  }

  // ---------------- Head: acts = h_j @ Wact^T + bact ----------------
  if (half == 0 && s < (long)Bn) {
    const float wa0 = Wact[0], wa1 = Wact[1], wa2 = Wact[2], wa3 = Wact[3];
    const float ba = bact[0];
    float* o = out + s * 7;
    #pragma unroll
    for (int t = 0; t < 7; ++t) {
      float acc = ba;
      acc += wa0 * (float)H[w][t][n][0];
      acc += wa1 * (float)H[w][t][n][1];
      acc += wa2 * (float)H[w][t][n][2];
      acc += wa3 * (float)H[w][t][n][3];
      o[t] = acc;
    }
  }
}

extern "C" void kernel_launch(void* const* d_in, const int* in_sizes, int n_in,
                              void* d_out, int out_size, void* d_ws, size_t ws_size,
                              hipStream_t stream) {
  const float* xp    = (const float*)d_in[0];
  const float* Wj    = (const float*)d_in[1];
  const float* bj    = (const float*)d_in[2];
  const float* Wm    = (const float*)d_in[3];
  const float* bm    = (const float*)d_in[4];
  const float* Wih_j = (const float*)d_in[5];
  const float* Whh_j = (const float*)d_in[6];
  const float* bih_j = (const float*)d_in[7];
  const float* bhh_j = (const float*)d_in[8];
  const float* Wih_m = (const float*)d_in[9];
  const float* Whh_m = (const float*)d_in[10];
  const float* bih_m = (const float*)d_in[11];
  const float* bhh_m = (const float*)d_in[12];
  const float* Wact  = (const float*)d_in[13];
  const float* bact  = (const float*)d_in[14];

  const int Bn = in_sizes[0] / 19;                 // 2,097,152
  const int blocks = (Bn + 127) / 128;             // 128 samples / block (8 waves x 16)

  aggreg_policy_wmma<<<blocks, 256, 0, stream>>>(
      xp, Wj, bj, Wm, bm, Wih_j, Whh_j, bih_j, bhh_j,
      Wih_m, Whh_m, bih_m, bhh_m, Wact, bact,
      (float*)d_out, Bn);
}
